// MoE_1984274891212
// MI455X (gfx1250) — matrix operand order repaired and checked by reading, not compiled, not run on hardware
//
#include <hip/hip_runtime.h>
#include <hip/hip_bf16.h>

// ---------------------------------------------------------------------------
// MoE layer for MI455X (gfx1250, wave32, WMMA).
//   H=2048 hidden, I=1408 per-expert inter, E=16 experts, top-K=4,
//   shared-expert inter ISH=2816, T = 4*2048 = 8192 tokens.
//
// Roofline: ~620 MB of weights stream once (NT-hinted so they do not evict
// the L2-resident x/h/out working set ~160 MB < 192 MB L2); 23.3 TB/s HBM
// => ~33 us floor. Compute (~2.6 TFLOP f16-WMMA) is noise next to that.
// GEMMs: v_wmma_f32_16x16x32_f16, f32 accumulation, f16 conversion in regs.
// x-tiles staged to LDS with global_load_async_to_lds_b128 (ASYNCcnt),
// double-buffered.
// ---------------------------------------------------------------------------

#define MOE_H   2048
#define MOE_I   1408
#define MOE_E   16
#define MOE_K   4
#define MOE_ISH 2816
#define MOE_T   8192

typedef __attribute__((ext_vector_type(16))) _Float16 v16h;
typedef __attribute__((ext_vector_type(8)))  float    v8f;
typedef __attribute__((ext_vector_type(4)))  float    v4f;

// A-tile staging in LDS: 128 rows x 32 K floats, padded row stride 36 floats
// (multiple of 4 for b128 alignment; 36 % 64 banks spreads rows).
#define ATILE_STRIDE 36
#define ATILE_FLOATS (128 * ATILE_STRIDE)

// ---------------------------------------------------------------------------
// CDNA5 async copy: global -> LDS, 16 bytes per lane, tracked by ASYNCcnt.
// ---------------------------------------------------------------------------
__device__ inline void async_copy_b128(unsigned lds_byte_off,
                                       unsigned long long gaddr) {
    asm volatile("global_load_async_to_lds_b128 %0, %1, off"
                 :: "v"(lds_byte_off), "v"(gaddr)
                 : "memory");
}

template <int N>
__device__ inline void wait_asynccnt() {
#if __has_builtin(__builtin_amdgcn_s_wait_asynccnt)
    __builtin_amdgcn_s_wait_asynccnt(N);
#else
    asm volatile("s_wait_asynccnt %0" :: "i"(N) : "memory");
#endif
}

// ---------------------------------------------------------------------------
// Fragment loaders (per-lane, wave32), WMMA f16 16x16x32 layouts (ISA 7.12.2):
// A (16x32 MxK): lanes 0-15: row=lane,     K = {k0..k0+7, k0+16..k0+23}
//                lanes16-31: row=lane-16,  K = {k0+8..15, k0+24..31}
// B (32x16 KxN): lane<16: col=lane, K = {k0..k0+7, k0+16..k0+23};
//                lane>=16: col=lane-16, K = {k0+8..15, k0+24..31}. B source is row-major [N,K].
// ---------------------------------------------------------------------------
template <typename SrcT>
__device__ inline v16h load_frag_a(const SrcT* __restrict__ src, int ld,
                                   int row_base, int k0) {
    const int lane = threadIdx.x & 31;
    const int m    = lane & 15;
    const int kb   = (lane >> 4) * 8;
    const SrcT* p  = src + (size_t)(row_base + m) * ld + (k0 + kb);
    v16h r;
#pragma unroll
    for (int i = 0; i < 8; ++i) r[i]     = (_Float16)p[i];
#pragma unroll
    for (int i = 0; i < 8; ++i) r[8 + i] = (_Float16)p[16 + i];
    return r;
}

// B fragment from fp32 weights, non-temporal (streamed once, keep out of L2).
__device__ inline v16h load_frag_b_nt(const float* __restrict__ src, int ld,
                                      int col_base, int k0) {
    const int lane = threadIdx.x & 31;
    const int n    = lane & 15;
    const int kh   = (lane >> 4) * 8;
    const v4f* p   = (const v4f*)(src + (size_t)(col_base + n) * ld + (k0 + kh));
    v4f c0 = __builtin_nontemporal_load(p);
    v4f c1 = __builtin_nontemporal_load(p + 1);
    v4f c2 = __builtin_nontemporal_load(p + 4);
    v4f c3 = __builtin_nontemporal_load(p + 5);
    v16h r;
#pragma unroll
    for (int i = 0; i < 4; ++i) r[i]      = (_Float16)c0[i];
#pragma unroll
    for (int i = 0; i < 4; ++i) r[4 + i]  = (_Float16)c1[i];
#pragma unroll
    for (int i = 0; i < 4; ++i) r[8 + i]  = (_Float16)c2[i];
#pragma unroll
    for (int i = 0; i < 4; ++i) r[12 + i] = (_Float16)c3[i];
    return r;
}

__device__ inline v8f wmma_f16(v16h a, v16h b, v8f c) {
    return __builtin_amdgcn_wmma_f32_16x16x32_f16(
        /*neg_a=*/false, a, /*neg_b=*/false, b,
        /*c_mod=*/(short)0, c, /*reuse_a=*/false, /*reuse_b=*/false);
}

// ---------------------------------------------------------------------------
// Router: one wave per token. Lane L owns expert (L&15) over K-half (L>>4).
// ---------------------------------------------------------------------------
__global__ void moe_router_kernel(const float* __restrict__ x,
                                  const float* __restrict__ rg,
                                  const float* __restrict__ ru,
                                  const float* __restrict__ escale,
                                  const float* __restrict__ ebias,
                                  float* __restrict__ coeff) {
    const int wavesPerBlock = blockDim.x >> 5;
    const int t    = blockIdx.x * wavesPerBlock + (threadIdx.x >> 5);
    const int lane = threadIdx.x & 31;
    const int e    = lane & 15;
    const int kh   = (lane >> 4) * (MOE_H / 2);

    const float* xp = x  + (size_t)t * MOE_H + kh;
    const float* gp = rg + (size_t)e * MOE_H + kh;
    const float* up = ru + (size_t)e * MOE_H + kh;

    float ag = 0.f, au = 0.f;
    for (int k = 0; k < MOE_H / 2; ++k) {
        const float xv = xp[k];
        ag = fmaf(xv, gp[k], ag);
        au = fmaf(xv, up[k], au);
    }
    ag += __shfl_xor(ag, 16);
    au += __shfl_xor(au, 16);

    const float silu  = ag / (1.f + __expf(-ag));
    const float score = fabsf(au * silu);

    float mx = score;
#pragma unroll
    for (int o = 1; o < 16; o <<= 1) mx = fmaxf(mx, __shfl_xor(mx, o, 16));
    const float ex = __expf(score - mx);
    float sum = ex;
#pragma unroll
    for (int o = 1; o < 16; o <<= 1) sum += __shfl_xor(sum, o, 16);
    const float prob = ex / sum;

    float biased = prob + ebias[e];
    int   sel    = 0;
    float b      = biased;
#pragma unroll
    for (int kk = 0; kk < MOE_K; ++kk) {
        float v  = b;
        int   ei = e;
#pragma unroll
        for (int o = 1; o < 16; o <<= 1) {
            const float v2 = __shfl_xor(v,  o, 16);
            const int   e2 = __shfl_xor(ei, o, 16);
            if (v2 > v || (v2 == v && e2 < ei)) { v = v2; ei = e2; }
        }
        if (e == ei) { sel = 1; b = -1e30f; }
    }
    if (lane < 16)
        coeff[(size_t)t * MOE_E + e] = sel ? (1.f + prob * escale[e]) : 0.f;
}

// ---------------------------------------------------------------------------
// Fused gate/up GEMM: h[t, n] = silu(x@Wg^T) * (x@Wu^T), K = H = 2048.
// Block: 256 threads = 8 waves (4 M x 2 N). Tile 128 x 64; wave 32x32.
// x tile (128x32 f32, 16 KB) staged via async-to-LDS, double-buffered.
// ---------------------------------------------------------------------------
__global__ void __launch_bounds__(256)
moe_gateup_kernel(const float* __restrict__ x,
                  const float* __restrict__ wg,
                  const float* __restrict__ wu,
                  _Float16* __restrict__ hout, int N) {
    __shared__ float atile[2][ATILE_FLOATS];

    const int tid    = threadIdx.x;
    const int wave   = tid >> 5;
    const int wave_m = wave & 3;
    const int wave_n = wave >> 2;
    const int blk_row = blockIdx.x * 128;                // block token base
    const int rowl    = wave_m * 32;                     // wave row (local)
    const int colb    = blockIdx.y * 64 + wave_n * 32;   // wave N base

    const unsigned abase     = (unsigned)(uintptr_t)&atile[0][0];
    const unsigned bufstride = ATILE_FLOATS * sizeof(float);

    const int lane = tid & 31;
    const int m    = lane & 15;
    const int kb   = (lane >> 4) * 8;

    // Stage issuance: 1024 x 16B chunks, 4 per thread, one K-step (32 floats).
    auto issue_stage = [&](int k0, unsigned lds_base) {
#pragma unroll
        for (int c = 0; c < 4; ++c) {
            const int ci  = c * 256 + tid;        // 0..1023
            const int row = ci >> 3;              // 0..127
            const int ko  = (ci & 7) * 4;         // 0..28
            const unsigned loff =
                lds_base + (unsigned)((row * ATILE_STRIDE + ko) * sizeof(float));
            const unsigned long long ga = (unsigned long long)(
                x + (size_t)(blk_row + row) * MOE_H + (k0 + ko));
            async_copy_b128(loff, ga);
        }
    };

    v8f accg[2][2], accu[2][2];
#pragma unroll
    for (int i = 0; i < 2; ++i)
#pragma unroll
        for (int j = 0; j < 2; ++j) { accg[i][j] = (v8f)0.f; accu[i][j] = (v8f)0.f; }

    constexpr int NSTEP = MOE_H / 32;            // 64
    issue_stage(0, abase);

    for (int s = 0; s < NSTEP; ++s) {
        const int k0 = s * 32;
        if (s + 1 < NSTEP) {
            issue_stage(k0 + 32, abase + (unsigned)((s + 1) & 1) * bufstride);
            wait_asynccnt<4>();                  // previous stage landed
        } else {
            wait_asynccnt<0>();
        }
        __syncthreads();                         // all waves' chunks visible

        const int buf = s & 1;
        v16h a[2], bg[2], bu[2];
#pragma unroll
        for (int i = 0; i < 2; ++i) {
            const int rl = rowl + 16 * i + m;
            const float* p = &atile[buf][rl * ATILE_STRIDE + kb];
            v16h r;
#pragma unroll
            for (int q = 0; q < 8; ++q) r[q]     = (_Float16)p[q];
#pragma unroll
            for (int q = 0; q < 8; ++q) r[8 + q] = (_Float16)p[16 + q];
            a[i] = r;
        }
#pragma unroll
        for (int j = 0; j < 2; ++j) {
            bg[j] = load_frag_b_nt(wg, MOE_H, colb + 16 * j, k0);
            bu[j] = load_frag_b_nt(wu, MOE_H, colb + 16 * j, k0);
        }
#pragma unroll
        for (int i = 0; i < 2; ++i)
#pragma unroll
            for (int j = 0; j < 2; ++j) {
                accg[i][j] = wmma_f16(a[i], bg[j], accg[i][j]);
                accu[i][j] = wmma_f16(a[i], bu[j], accu[i][j]);
            }
        __syncthreads();                         // done reading buf
    }

    // Epilogue: C layout -> VGPR r: M = r + (lane>=16 ? 8:0), N = lane&15.
    const int hi = (lane >> 4) * 8;
#pragma unroll
    for (int i = 0; i < 2; ++i)
#pragma unroll
        for (int j = 0; j < 2; ++j) {
            const int col = colb + 16 * j + m;
#pragma unroll
            for (int r = 0; r < 8; ++r) {
                const int row = blk_row + rowl + 16 * i + hi + r;
                const float g = accg[i][j][r];
                const float u = accu[i][j][r];
                const float sv = g / (1.f + __expf(-g));
                hout[(size_t)row * N + col] = (_Float16)(sv * u);
            }
        }
}

// ---------------------------------------------------------------------------
// Down projection: out[t, n] (+)= (h @ Wd^T)[t, n] * coeff[t, e]
// h: [T, Kdim] f16 (L2-resident workspace), Wd: [H, Kdim] f32 NT, N = H.
// ---------------------------------------------------------------------------
__global__ void __launch_bounds__(256)
moe_down_kernel(const _Float16* __restrict__ hin,
                const float* __restrict__ wd,
                float* __restrict__ out,
                const float* __restrict__ coeff,
                int e, int Kdim, int use_coeff, int accumulate) {
    const int wave   = threadIdx.x >> 5;
    const int wave_m = wave & 3;
    const int wave_n = wave >> 2;
    const int rowb   = blockIdx.x * 128 + wave_m * 32;
    const int colb   = blockIdx.y * 64  + wave_n * 32;

    v8f acc[2][2];
#pragma unroll
    for (int i = 0; i < 2; ++i)
#pragma unroll
        for (int j = 0; j < 2; ++j) acc[i][j] = (v8f)0.f;

    for (int k0 = 0; k0 < Kdim; k0 += 32) {
        v16h a[2], b[2];
#pragma unroll
        for (int i = 0; i < 2; ++i) a[i] = load_frag_a(hin, Kdim, rowb + 16 * i, k0);
#pragma unroll
        for (int j = 0; j < 2; ++j) b[j] = load_frag_b_nt(wd, Kdim, colb + 16 * j, k0);
#pragma unroll
        for (int i = 0; i < 2; ++i)
#pragma unroll
            for (int j = 0; j < 2; ++j)
                acc[i][j] = wmma_f16(a[i], b[j], acc[i][j]);
    }

    const int lane = threadIdx.x & 31;
    const int hi   = (lane >> 4) * 8;
    const int nn   = lane & 15;
#pragma unroll
    for (int i = 0; i < 2; ++i)
#pragma unroll
        for (int j = 0; j < 2; ++j) {
            const int col = colb + 16 * j + nn;
#pragma unroll
            for (int r = 0; r < 8; ++r) {
                const int row = rowb + 16 * i + hi + r;
                const float c = use_coeff ? coeff[(size_t)row * MOE_E + e] : 1.f;
                const float v = acc[i][j][r] * c;
                const size_t idx = (size_t)row * MOE_H + col;
                out[idx] = accumulate ? (out[idx] + v) : v;
            }
        }
}

// ---------------------------------------------------------------------------
// Launch. Inputs (setup_inputs order):
//  0:x 1:rg_w 2:ru_w 3:extra_scale 4:extra_bias 5:Wg 6:Wu 7:Wd 8:Sg 9:Su 10:Sd
// Workspace: coeff[T*E] f32, then hbuf[T*ISH] f16 (~46 MB, L2-resident).
// Experts processed sequentially so x/hbuf/out stay hot in the 192 MB L2;
// weight streams are NT-hinted so they do not evict them.
// ---------------------------------------------------------------------------
extern "C" void kernel_launch(void* const* d_in, const int* in_sizes, int n_in,
                              void* d_out, int out_size, void* d_ws, size_t ws_size,
                              hipStream_t stream) {
    const float* x      = (const float*)d_in[0];
    const float* rg     = (const float*)d_in[1];
    const float* ru     = (const float*)d_in[2];
    const float* escale = (const float*)d_in[3];
    const float* ebias  = (const float*)d_in[4];
    const float* Wg     = (const float*)d_in[5];
    const float* Wu     = (const float*)d_in[6];
    const float* Wd     = (const float*)d_in[7];
    const float* Sg     = (const float*)d_in[8];
    const float* Su     = (const float*)d_in[9];
    const float* Sd     = (const float*)d_in[10];
    float* out = (float*)d_out;

    float*    coeff = (float*)d_ws;
    _Float16* hbuf  = (_Float16*)((char*)d_ws + (size_t)MOE_T * MOE_E * sizeof(float));

    // 1) Router -> coeff[T, E]
    moe_router_kernel<<<MOE_T / 8, 256, 0, stream>>>(x, rg, ru, escale, ebias, coeff);

    // 2) Shared expert (writes out, no coeff)
    moe_gateup_kernel<<<dim3(MOE_T / 128, MOE_ISH / 64), 256, 0, stream>>>(
        x, Sg, Su, hbuf, MOE_ISH);
    moe_down_kernel<<<dim3(MOE_T / 128, MOE_H / 64), 256, 0, stream>>>(
        hbuf, Sd, out, nullptr, 0, MOE_ISH, /*use_coeff=*/0, /*accumulate=*/0);

    // 3) Routed experts, sequential for L2 reuse of x and hbuf
    for (int e = 0; e < MOE_E; ++e) {
        moe_gateup_kernel<<<dim3(MOE_T / 128, MOE_I / 64), 256, 0, stream>>>(
            x, Wg + (size_t)e * MOE_I * MOE_H, Wu + (size_t)e * MOE_I * MOE_H,
            hbuf, MOE_I);
        moe_down_kernel<<<dim3(MOE_T / 128, MOE_H / 64), 256, 0, stream>>>(
            hbuf, Wd + (size_t)e * MOE_H * MOE_I, out, coeff, e, MOE_I,
            /*use_coeff=*/1, /*accumulate=*/1);
    }
}
